// MoELayer_33260226740433
// MI455X (gfx1250) — compile-verified, matrix operand order checked by reading, not run on hardware
//
#include <hip/hip_runtime.h>
#include <hip/hip_bf16.h>
#include <math.h>
#include <stdint.h>

typedef __bf16 bf16_t;
typedef __attribute__((ext_vector_type(16))) __bf16 v16bf;
typedef __attribute__((ext_vector_type(8)))  __bf16 v8bf;
typedef __attribute__((ext_vector_type(8)))  float  v8f;
typedef __attribute__((ext_vector_type(4)))  int    v4i;

#define NTOK 4096
#define DDIM 768
#define FDIM 3072
#define NEXP 8
#define BM 128
#define BN 128
#define BK 32
#define LDSPAD 8   // row stride BK+8 bf16 = 80 B (16-B aligned)

union FragBF { v16bf v; v8bf h[2]; };

// ---------------------------------------------------------------------------
// gfx1250 async memory->LDS path (ASYNCcnt), guarded so compilation can't
// regress if the builtin is unavailable on this toolchain.
// Parameter types per compiler diagnostic: (global int4*, local int4*, imm, imm)
// ---------------------------------------------------------------------------
#if __has_builtin(__builtin_amdgcn_global_load_async_to_lds_b128)
#define USE_ASYNC_LDS 1
typedef __attribute__((address_space(1))) v4i gbl_v4i;
typedef __attribute__((address_space(3))) v4i lds_v4i;
#endif

__device__ inline void async_copy_b128(void* lds_dst, const void* gsrc)
{
#ifdef USE_ASYNC_LDS
    __builtin_amdgcn_global_load_async_to_lds_b128(
        (gbl_v4i*)(uintptr_t)gsrc,
        (lds_v4i*)(uint32_t)(uintptr_t)lds_dst,
        0, 0);
#else
    *(uint4*)lds_dst = *(const uint4*)gsrc;
#endif
}

__device__ inline void async_wait()
{
#ifdef USE_ASYNC_LDS
#if __has_builtin(__builtin_amdgcn_s_wait_asynccnt)
    __builtin_amdgcn_s_wait_asynccnt(0);
#else
    asm volatile("s_wait_asynccnt 0" ::: "memory");
#endif
#endif
}

// ---------------------------------------------------------------------------
// Kernel 1: gating — one wave per token. Exact softmax, top-2, renormalize.
// ---------------------------------------------------------------------------
__global__ __launch_bounds__(256)
void gate_topk_kernel(const float* __restrict__ x,
                      const float* __restrict__ gate_w,   // [D][E]
                      const float* __restrict__ gate_b,   // [E]
                      float* __restrict__ combine,        // [N][E]
                      int* __restrict__ top2)             // [N][2]
{
    const int wave = threadIdx.x >> 5;
    const int lane = threadIdx.x & 31;
    const int n = blockIdx.x * 8 + wave;
    if (n >= NTOK) return;

    float acc[NEXP];
#pragma unroll
    for (int e = 0; e < NEXP; ++e) acc[e] = 0.0f;

    const float* xr = x + (size_t)n * DDIM;
    for (int d = lane; d < DDIM; d += 32) {
        const float xv = xr[d];
        const float* gw = gate_w + (size_t)d * NEXP;
#pragma unroll
        for (int e = 0; e < NEXP; ++e) acc[e] += xv * gw[e];
    }
#pragma unroll
    for (int e = 0; e < NEXP; ++e) {
        float v = acc[e];
#pragma unroll
        for (int off = 16; off > 0; off >>= 1) v += __shfl_xor(v, off, 32);
        acc[e] = v;
    }
    if (lane == 0) {
        float logit[NEXP], g[NEXP];
        float mx = -1e30f;
#pragma unroll
        for (int e = 0; e < NEXP; ++e) { logit[e] = acc[e] + gate_b[e]; mx = fmaxf(mx, logit[e]); }
        float s = 0.0f;
#pragma unroll
        for (int e = 0; e < NEXP; ++e) { g[e] = expf(logit[e] - mx); s += g[e]; }
        const float inv = 1.0f / s;
#pragma unroll
        for (int e = 0; e < NEXP; ++e) g[e] *= inv;
        int i1 = 0;
#pragma unroll
        for (int e = 1; e < NEXP; ++e) if (g[e] > g[i1]) i1 = e;
        int i2 = (i1 == 0) ? 1 : 0;
#pragma unroll
        for (int e = 0; e < NEXP; ++e) if (e != i1 && g[e] > g[i2]) i2 = e;
        const float s2 = g[i1] + g[i2] + 1e-9f;
        float* crow = combine + (size_t)n * NEXP;
#pragma unroll
        for (int e = 0; e < NEXP; ++e) crow[e] = 0.0f;
        crow[i1] = g[i1] / s2;
        crow[i2] = g[i2] / s2;
        top2[2 * n + 0] = i1;
        top2[2 * n + 1] = i2;
    }
}

// ---------------------------------------------------------------------------
// Kernel 2: deterministic per-expert token lists (token order preserved).
// ---------------------------------------------------------------------------
__global__ void build_lists_kernel(const int* __restrict__ top2,
                                   int* __restrict__ counts,
                                   int* __restrict__ tok_idx)   // [E][N]
{
    const int e = threadIdx.x;
    if (e >= NEXP) return;
    int c = 0;
    for (int n = 0; n < NTOK; ++n) {
        if (top2[2 * n] == e || top2[2 * n + 1] == e) tok_idx[e * NTOK + c++] = n;
    }
    counts[e] = c;
}

__global__ void zero_kernel(float* __restrict__ p, int n)
{
    const int i = blockIdx.x * blockDim.x + threadIdx.x;
    if (i < n) p[i] = 0.0f;
}

// ---------------------------------------------------------------------------
// Fragment load: 16-bit WMMA A/B layout. Lane l = half*16 + m:
//   elements 0..7  = row[half*8 + 0..7]    (one 16-byte LDS chunk)
//   elements 8..15 = row[16 + half*8 + 0..7]
// ---------------------------------------------------------------------------
__device__ inline v16bf load_frag_row(const bf16_t* row, int half)
{
    const v8bf* r = (const v8bf*)row;
    FragBF f;
    f.h[0] = r[half];
    f.h[1] = r[2 + half];
    return f.v;
}

// convert 16 fp32 -> bf16 in registers, store as 2x ds_store_b128
__device__ inline void cvt_store16(bf16_t* dst, const float* s)
{
    union { bf16_t h[16]; uint4 q[2]; } u;
#pragma unroll
    for (int i = 0; i < 16; ++i) u.h[i] = (bf16_t)s[i];
    *(uint4*)(dst)     = u.q[0];
    *(uint4*)(dst + 8) = u.q[1];
}

// ---------------------------------------------------------------------------
// Kernel 3: h = gelu( X[gather] @ W1[e] + b1[e] )   grid=(F/BN, N/BM), 8 waves
// Double-buffered LDS; global prefetch of step k+1 overlaps WMMA of step k.
// ---------------------------------------------------------------------------
__global__ __launch_bounds__(256)
void moe_gemm1_kernel(const float* __restrict__ x,        // [N][D]
                      const float* __restrict__ w1,       // [E][D][F]
                      const float* __restrict__ b1,       // [E][F]
                      const int* __restrict__ counts,
                      const int* __restrict__ tok_idx,    // [E][N]
                      bf16_t* __restrict__ h,             // [N][F]
                      int e)
{
    const int Ne = counts[e];
    const int m0 = blockIdx.y * BM;
    if (m0 >= Ne) return;
    const int n0 = blockIdx.x * BN;

    __shared__ __align__(16) bf16_t As[2][BM][BK + LDSPAD];
    __shared__ __align__(16) bf16_t BsT[2][BN][BK + LDSPAD];
    __shared__ int toks[BM];

    const int tid = threadIdx.x;
    if (tid < BM) {
        const int slot = m0 + tid;
        toks[tid] = (slot < Ne) ? tok_idx[e * NTOK + slot] : -1;
    }
    __syncthreads();

    // staging roles
    const int ar = tid >> 1;            // A row 0..127 (2 threads/row)
    const int ac = (tid & 1) * 16;      // A col chunk: 0 or 16
    const int atok = toks[ar];
    const int bc = tid & 127;           // B column within tile (lane-coalesced)
    const int bkb = (tid >> 7) * 16;    // B k-chunk base: 0 or 16

    // zero-fill padded gather rows once (both buffers)
    if (atok < 0) {
        const uint4 z = {0, 0, 0, 0};
#pragma unroll
        for (int b = 0; b < 2; ++b) {
            *(uint4*)&As[b][ar][ac]     = z;
            *(uint4*)&As[b][ar][ac + 8] = z;
        }
    }

    const float* W = w1 + (size_t)e * DDIM * FDIM;
    const float* xrow = (atok >= 0) ? (x + (size_t)atok * DDIM) : x;

    const int wave = tid >> 5;
    const int lane = tid & 31;
    const int half = lane >> 4;
    const int ml   = lane & 15;

    v8f acc[8];
#pragma unroll
    for (int t = 0; t < 8; ++t) acc[t] = (v8f)(0.0f);

    float a_reg[16], b_reg[16];

    // ---- prologue: fetch tile 0, stage into buffer 0 ----
    if (atok >= 0) {
#pragma unroll
        for (int j = 0; j < 16; j += 4)
            *(float4*)&a_reg[j] = *(const float4*)(xrow + ac + j);
    }
    {
        const float* bsrc = W + (size_t)bkb * FDIM + n0 + bc;
#pragma unroll
        for (int j = 0; j < 16; ++j) b_reg[j] = bsrc[(size_t)j * FDIM];
    }
    if (atok >= 0) cvt_store16(&As[0][ar][ac], a_reg);
    cvt_store16(&BsT[0][bc][bkb], b_reg);
    __syncthreads();

    // ---- main loop: prefetch k+1, compute k, stage k+1, barrier ----
    int p = 0;
    for (int step = 0; step < DDIM / BK; ++step) {
        const int knext = (step + 1) * BK;
        if (knext < DDIM) {
            if (atok >= 0) {
#pragma unroll
                for (int j = 0; j < 16; j += 4)
                    *(float4*)&a_reg[j] = *(const float4*)(xrow + knext + ac + j);
            }
            const float* bsrc = W + (size_t)(knext + bkb) * FDIM + n0 + bc;
#pragma unroll
            for (int j = 0; j < 16; ++j) b_reg[j] = bsrc[(size_t)j * FDIM];
        }

        const v16bf a = load_frag_row(&As[p][wave * 16 + ml][0], half);
#pragma unroll
        for (int t = 0; t < 8; ++t) {
            const v16bf b = load_frag_row(&BsT[p][t * 16 + ml][0], half);
            acc[t] = __builtin_amdgcn_wmma_f32_16x16x32_bf16(
                false, a, false, b, (short)0, acc[t], false, false);
        }

        if (knext < DDIM) {
            if (atok >= 0) cvt_store16(&As[1 - p][ar][ac], a_reg);
            cvt_store16(&BsT[1 - p][bc][bkb], b_reg);
        }
        __syncthreads();
        p ^= 1;
    }

    // ---- epilogue: bias + exact GELU, store bf16 h ----
#pragma unroll
    for (int t = 0; t < 8; ++t) {
        const int ncol = n0 + t * 16 + ml;
        const float bias = b1[(size_t)e * FDIM + ncol];
#pragma unroll
        for (int i = 0; i < 8; ++i) {
            const int slot = m0 + wave * 16 + half * 8 + i;
            if (slot < Ne) {
                const float v = acc[t][i] + bias;
                const float g = 0.5f * v * (1.0f + erff(v * 0.70710678118654752f));
                h[(size_t)slot * FDIM + ncol] = (bf16_t)g;
            }
        }
    }
}

// ---------------------------------------------------------------------------
// Kernel 4: out[tok] += combine[tok,e]*(h @ W2[e] + b2[e])  grid=(D/BN, N/BM)
// A tile (already bf16) uses GLOBAL_LOAD_ASYNC_TO_LDS_B128 (ASYNCcnt) so the
// memory->LDS DMA overlaps the WMMA loop; B tile converts fp32->bf16 via regs.
// ---------------------------------------------------------------------------
__global__ __launch_bounds__(256)
void moe_gemm2_kernel(const bf16_t* __restrict__ h,       // [N][F]
                      const float* __restrict__ w2,       // [E][F][D]
                      const float* __restrict__ b2,       // [E][D]
                      const float* __restrict__ combine,  // [N][E]
                      const int* __restrict__ counts,
                      const int* __restrict__ tok_idx,    // [E][N]
                      float* __restrict__ out,            // [N][D]
                      int e)
{
    const int Ne = counts[e];
    const int m0 = blockIdx.y * BM;
    if (m0 >= Ne) return;
    const int n0 = blockIdx.x * BN;

    __shared__ __align__(16) bf16_t As[2][BM][BK + LDSPAD];
    __shared__ __align__(16) bf16_t BsT[2][BN][BK + LDSPAD];
    __shared__ int toks[BM];

    const int tid = threadIdx.x;
    if (tid < BM) {
        const int slot = m0 + tid;
        toks[tid] = (slot < Ne) ? tok_idx[e * NTOK + slot] : -1;
    }
    __syncthreads();

    const int ar = tid >> 1;
    const int ac = (tid & 1) * 16;
    const int slotA = m0 + ar;
    const bool arow_ok = (slotA < Ne);
    const bf16_t* hrow = h + (size_t)(arow_ok ? slotA : 0) * FDIM;
    const int bc = tid & 127;
    const int bkb = (tid >> 7) * 16;

    if (!arow_ok) {
        const uint4 z = {0, 0, 0, 0};
#pragma unroll
        for (int b = 0; b < 2; ++b) {
            *(uint4*)&As[b][ar][ac]     = z;
            *(uint4*)&As[b][ar][ac + 8] = z;
        }
    }

    const float* W = w2 + (size_t)e * FDIM * DDIM;

    const int wave = tid >> 5;
    const int lane = tid & 31;
    const int half = lane >> 4;
    const int ml   = lane & 15;

    v8f acc[8];
#pragma unroll
    for (int t = 0; t < 8; ++t) acc[t] = (v8f)(0.0f);

    float b_reg[16];

    // ---- prologue: tile 0 ----
    if (arow_ok) {
        async_copy_b128(&As[0][ar][ac],     hrow + ac);
        async_copy_b128(&As[0][ar][ac + 8], hrow + ac + 8);
    }
    {
        const float* bsrc = W + (size_t)bkb * DDIM + n0 + bc;
#pragma unroll
        for (int j = 0; j < 16; ++j) b_reg[j] = bsrc[(size_t)j * DDIM];
    }
    cvt_store16(&BsT[0][bc][bkb], b_reg);
    async_wait();
    __syncthreads();

    // ---- main loop ----
    int p = 0;
    for (int step = 0; step < FDIM / BK; ++step) {
        const int knext = (step + 1) * BK;
        if (knext < FDIM) {
            if (arow_ok) {
                async_copy_b128(&As[1 - p][ar][ac],     hrow + knext + ac);
                async_copy_b128(&As[1 - p][ar][ac + 8], hrow + knext + ac + 8);
            }
            const float* bsrc = W + (size_t)(knext + bkb) * DDIM + n0 + bc;
#pragma unroll
            for (int j = 0; j < 16; ++j) b_reg[j] = bsrc[(size_t)j * DDIM];
        }

        const v16bf a = load_frag_row(&As[p][wave * 16 + ml][0], half);
#pragma unroll
        for (int t = 0; t < 8; ++t) {
            const v16bf b = load_frag_row(&BsT[p][t * 16 + ml][0], half);
            acc[t] = __builtin_amdgcn_wmma_f32_16x16x32_bf16(
                false, a, false, b, (short)0, acc[t], false, false);
        }

        if (knext < FDIM) cvt_store16(&BsT[1 - p][bc][bkb], b_reg);
        async_wait();
        __syncthreads();
        p ^= 1;
    }

    // ---- epilogue: scale by combine weight, accumulate into out ----
#pragma unroll
    for (int t = 0; t < 8; ++t) {
        const int ncol = n0 + t * 16 + ml;
        const float bias = b2[(size_t)e * DDIM + ncol];
#pragma unroll
        for (int i = 0; i < 8; ++i) {
            const int row  = wave * 16 + half * 8 + i;
            const int slot = m0 + row;
            if (slot < Ne) {
                const int tok = toks[row];
                const float cw = combine[(size_t)tok * NEXP + e];
                float* dst = out + (size_t)tok * DDIM + ncol;
                *dst = *dst + cw * (acc[t][i] + bias);
            }
        }
    }
}

// ---------------------------------------------------------------------------
extern "C" void kernel_launch(void* const* d_in, const int* in_sizes, int n_in,
                              void* d_out, int out_size, void* d_ws, size_t ws_size,
                              hipStream_t stream)
{
    const float* x      = (const float*)d_in[0];
    const float* gate_w = (const float*)d_in[1];
    const float* gate_b = (const float*)d_in[2];
    const float* w1     = (const float*)d_in[3];
    const float* b1     = (const float*)d_in[4];
    const float* w2     = (const float*)d_in[5];
    const float* b2     = (const float*)d_in[6];
    float* out = (float*)d_out;

    // workspace layout
    char* ws = (char*)d_ws;
    float* combine = (float*)ws;           ws += (size_t)NTOK * NEXP * sizeof(float); // 128 KB
    int*   top2    = (int*)ws;             ws += (size_t)NTOK * 2 * sizeof(int);      // 32 KB
    int*   counts  = (int*)ws;             ws += 256;
    int*   tok_idx = (int*)ws;             ws += (size_t)NEXP * NTOK * sizeof(int);   // 128 KB
    bf16_t* h      = (bf16_t*)ws;          // N*F bf16 = 24 MB

    gate_topk_kernel<<<NTOK / 8, 256, 0, stream>>>(x, gate_w, gate_b, combine, top2);
    build_lists_kernel<<<1, 32, 0, stream>>>(top2, counts, tok_idx);
    {
        const int total = NTOK * DDIM;
        zero_kernel<<<(total + 255) / 256, 256, 0, stream>>>(out, total);
    }
    for (int e = 0; e < NEXP; ++e) {
        moe_gemm1_kernel<<<dim3(FDIM / BN, NTOK / BM), 256, 0, stream>>>(
            x, w1, b1, counts, tok_idx, h, e);
        moe_gemm2_kernel<<<dim3(DDIM / BN, NTOK / BM), 256, 0, stream>>>(
            h, w2, b2, combine, counts, tok_idx, out, e);
    }
}
